// SCConv_19885698580511
// MI455X (gfx1250) — compile-verified, matrix operand order checked by reading, not run on hardware
//
#include <hip/hip_runtime.h>
#include <hip/hip_bf16.h>
#include <math.h>

// ---------------------------------------------------------------------------
// CondConv (per-sample expert-mixed 3x3 conv) for MI455X / gfx1250.
// bf16 WMMA (V_WMMA_F32_16X16X32_BF16) implicit GEMM, f32 accumulate.
// Wave tile 32x32 (2 A-frags x 2 B-frags -> 4 WMMA per 8 ds_load_b128).
// ---------------------------------------------------------------------------

typedef __attribute__((ext_vector_type(16))) __bf16 v16bf;
typedef __attribute__((ext_vector_type(8)))  float  v8f;

#define BATCH 32
#define CIN   256
#define COUT  256
#define HW    56
#define NPIX  (HW * HW)      // 3136
#define NEXP  3
#define RED   16
#define NTAP  9

// conv tiling
#define MT    64             // c_out tile per block
#define TS    8              // spatial tile side (8x8 = 64 pixels)
#define KC    32             // c_in chunk per K-step
#define LSTR  40             // padded ci stride in LDS (ushorts): 80B rows, 16B aligned
#define CTHREADS 128         // 4 waves; each wave owns a 32x32 quadrant

// POD 16-byte chunk (HIP's uint4 has ctors -> avoid)
struct alignas(16) U4 { unsigned int x, y, z, w; };
struct U8 { U4 lo, hi; };    // 32 bytes == v16bf

static __device__ __forceinline__ unsigned short f2bf(float f) {
    unsigned int u = __float_as_uint(f);
    u += 0x7FFFu + ((u >> 16) & 1u);   // round to nearest even
    return (unsigned short)(u >> 16);
}

static __device__ __forceinline__ v16bf ld_frag(const unsigned short* p) {
    U8 u; u.lo = *(const U4*)p; u.hi = *(const U4*)(p + 16);
    return __builtin_bit_cast(v16bf, u);
}

// ---------------------------------------------------------------------------
// Kernel 1: routing  ->  rw[b][e] = sigmoid(relu(fc(relu(conv1x1(pool4(x))))))
// ---------------------------------------------------------------------------
__global__ __launch_bounds__(256) void route_kernel(
    const float* __restrict__ x, const float* __restrict__ rcw,
    const float* __restrict__ rcb, const float* __restrict__ fcw,
    const float* __restrict__ fcb, float* __restrict__ rw)
{
    __shared__ float pooled[CIN * 16];   // [c][bin]
    __shared__ float rflat[RED * 16];
    const int b = blockIdx.x, t = threadIdx.x;

    // Phase 1: exact 14x14 block means, one channel per thread.
    const float* xp = x + ((size_t)b * CIN + t) * NPIX;
    const float inv = 1.0f / 196.0f;
    for (int by = 0; by < 4; ++by) {
        float a0 = 0.f, a1 = 0.f, a2 = 0.f, a3 = 0.f;
        for (int y = by * 14; y < by * 14 + 14; ++y) {
            const float* row = xp + y * HW;
            float s0 = 0.f, s1 = 0.f, s2 = 0.f, s3 = 0.f;
            #pragma unroll
            for (int k = 0; k < 14; ++k) {
                s0 += row[k]; s1 += row[14 + k]; s2 += row[28 + k]; s3 += row[42 + k];
            }
            a0 += s0; a1 += s1; a2 += s2; a3 += s3;
        }
        pooled[t * 16 + by * 4 + 0] = a0 * inv;
        pooled[t * 16 + by * 4 + 1] = a1 * inv;
        pooled[t * 16 + by * 4 + 2] = a2 * inv;
        pooled[t * 16 + by * 4 + 3] = a3 * inv;
    }
    __syncthreads();

    // Phase 2: 1x1 conv to red=16 + relu; thread -> (rr, bin)
    {
        const int rr = t >> 4, bin = t & 15;
        float s = 0.f;
        for (int c = 0; c < CIN; ++c) s += pooled[c * 16 + bin] * rcw[rr * CIN + c];
        s += rcb[rr];
        rflat[rr * 16 + bin] = fmaxf(s, 0.f);
    }
    __syncthreads();

    // Phase 3: fc (3x256) + relu + sigmoid
    if (t < NEXP) {
        float acc = fcb[t];
        for (int j = 0; j < RED * 16; ++j) acc += rflat[j] * fcw[t * (RED * 16) + j];
        acc = fmaxf(acc, 0.f);
        rw[b * NEXP + t] = 1.0f / (1.0f + __expf(-acc));
    }
}

// ---------------------------------------------------------------------------
// Kernel 2: expert mixing -> cwb[b][tap][co][ci] in bf16 (coalesced writes;
// w_expert is 7MB -> fully L2-resident across all 32 samples).
// ---------------------------------------------------------------------------
__global__ __launch_bounds__(256) void mix_kernel(
    const float* __restrict__ we, const float* __restrict__ rw,
    unsigned short* __restrict__ cwb)
{
    const int ci  = threadIdx.x;
    const int co  = blockIdx.x;
    const int tap = blockIdx.y;
    const int b   = blockIdx.z;
    const float r0 = rw[b * NEXP + 0];
    const float r1 = rw[b * NEXP + 1];
    const float r2 = rw[b * NEXP + 2];
    const size_t estride = (size_t)COUT * CIN * NTAP;
    const size_t base = ((size_t)co * CIN + ci) * NTAP + tap;
    const float v = r0 * we[base] + r1 * we[base + estride] + r2 * we[base + 2 * estride];
    cwb[(((size_t)b * NTAP + tap) * COUT + co) * CIN + ci] = f2bf(v);
}

// ---------------------------------------------------------------------------
// Kernel 3: implicit-GEMM conv via bf16 WMMA.
// Block: 64 c_out x (8x8) pixels, 128 threads = 4 waves, wave tile 32x32.
// K loop: 8 chunks of 32 c_in, 9 taps each; 4 WMMA per 8 ds_load_b128.
// ---------------------------------------------------------------------------
__global__ __launch_bounds__(CTHREADS) void conv_kernel(
    const float* __restrict__ x, const unsigned short* __restrict__ cwb,
    float* __restrict__ out)
{
    __shared__ unsigned short lds_w[NTAP * MT * LSTR]; // [tap][co][ci pad40] 46080B
    __shared__ unsigned short lds_x[100 * LSTR];       // [halo pix][ci pad40] 8000B

    const int tid = threadIdx.x;
    const int b   = blockIdx.z;
    const int cot = blockIdx.y * MT;
    const int ty  = (blockIdx.x / 7) * TS;
    const int tx  = (blockIdx.x % 7) * TS;

    const unsigned short* cwb_b = cwb + (size_t)b * NTAP * COUT * CIN;
    const float* xb = x + (size_t)b * CIN * NPIX;

    const int lane = tid & 31;
    const int wv   = tid >> 5;                 // 4 waves
    const int moff = (wv >> 1) * 32;           // wave M offset (0 or 32)
    const int noff = (wv & 1) * 32;            // wave N offset (0 or 32)
    const int koff = (lane >> 4) * 8;          // ISA 16-bit A/B lane K offset
    const int mrow = lane & 15;                // A row / B col within fragment

    v8f c00 = {0.f,0.f,0.f,0.f,0.f,0.f,0.f,0.f};
    v8f c01 = {0.f,0.f,0.f,0.f,0.f,0.f,0.f,0.f};
    v8f c10 = {0.f,0.f,0.f,0.f,0.f,0.f,0.f,0.f};
    v8f c11 = {0.f,0.f,0.f,0.f,0.f,0.f,0.f,0.f};

    for (int cc = 0; cc < CIN / KC; ++cc) {
        const int ci0 = cc * KC;
        __syncthreads();

        // Stage mixed weights: [tap][co 64][ci 32] bf16, 2304 16B chunks.
        #pragma unroll
        for (int i = 0; i < 18; ++i) {
            const int q      = i * CTHREADS + tid;
            const int cipart = q & 3;           // 4 x 8 bf16 per (tap,co)
            const int co     = (q >> 2) & 63;
            const int tap    = q >> 8;
            const U4 v = *(const U4*)(cwb_b + ((size_t)(tap * COUT + cot + co)) * CIN
                                      + ci0 + cipart * 8);
            *(U4*)&lds_w[(tap * MT + co) * LSTR + cipart * 8] = v;
        }
        // Stage haloed x patch 10x10 x 32ci, f32 -> bf16 on the fly (3200 elems).
        #pragma unroll
        for (int i = 0; i < 25; ++i) {
            const int idx = i * CTHREADS + tid;
            const int ci  = idx / 100;
            const int pix = idx - ci * 100;
            const int hy  = pix / 10, hx = pix - hy * 10;
            const int gy  = ty - 1 + hy, gx = tx - 1 + hx;
            float v = 0.f;
            if ((unsigned)gy < HW && (unsigned)gx < HW)
                v = xb[(size_t)(ci0 + ci) * NPIX + gy * HW + gx];
            lds_x[pix * LSTR + ci] = f2bf(v);
        }
        __syncthreads();

        // 9 taps x (2 M-frags x 2 N-frags) WMMAs, K=32 per step.
        #pragma unroll
        for (int tap = 0; tap < 9; ++tap) {
            const int kh = tap / 3, kw = tap % 3;
            // B fragments: column = output pixel, K = ci (two 8-elem runs/lane)
            const int n0  = noff + mrow;
            const int n1  = n0 + 16;
            const int hp0 = ((n0 >> 3) + kh) * 10 + ((n0 & 7) + kw);
            const int hp1 = ((n1 >> 3) + kh) * 10 + ((n1 & 7) + kw);
            const v16bf b0 = ld_frag(&lds_x[hp0 * LSTR + koff]);
            const v16bf b1 = ld_frag(&lds_x[hp1 * LSTR + koff]);
            // A fragments: row = c_out
            const v16bf a0 = ld_frag(&lds_w[(tap * MT + moff + mrow) * LSTR + koff]);
            const v16bf a1 = ld_frag(&lds_w[(tap * MT + moff + 16 + mrow) * LSTR + koff]);

            c00 = __builtin_amdgcn_wmma_f32_16x16x32_bf16(
                      false, a0, false, b0, (short)0, c00, false, false);
            c01 = __builtin_amdgcn_wmma_f32_16x16x32_bf16(
                      false, a0, false, b1, (short)0, c01, false, false);
            c10 = __builtin_amdgcn_wmma_f32_16x16x32_bf16(
                      false, a1, false, b0, (short)0, c10, false, false);
            c11 = __builtin_amdgcn_wmma_f32_16x16x32_bf16(
                      false, a1, false, b1, (short)0, c11, false, false);
        }
    }

    // Epilogue: D layout — M = vgpr + 8*(lane>=16), N = lane&15.
    const int corow = (lane >> 4) << 3;
    float* outb = out + (size_t)b * COUT * NPIX;
    #pragma unroll
    for (int g = 0; g < 2; ++g) {
        const int n  = noff + g * 16 + (lane & 15);
        const int py = n >> 3, px = n & 7;
        float* ob = outb + (size_t)((ty + py) * HW + tx + px);
        const v8f& lo = g ? c01 : c00;
        const v8f& hi = g ? c11 : c10;
        #pragma unroll
        for (int r = 0; r < 8; ++r) {
            const int co0 = cot + moff + corow + r;
            ob[(size_t)co0 * NPIX]        = lo[r];
            ob[(size_t)(co0 + 16) * NPIX] = hi[r];
        }
    }
}

// ---------------------------------------------------------------------------
extern "C" void kernel_launch(void* const* d_in, const int* in_sizes, int n_in,
                              void* d_out, int out_size, void* d_ws, size_t ws_size,
                              hipStream_t stream) {
    const float* x   = (const float*)d_in[0];   // [32,256,56,56]
    const float* we  = (const float*)d_in[1];   // [3,256,256,3,3]
    const float* rcw = (const float*)d_in[2];   // [16,256]
    const float* rcb = (const float*)d_in[3];   // [16]
    const float* fcw = (const float*)d_in[4];   // [3,256]
    const float* fcb = (const float*)d_in[5];   // [3]
    float* out = (float*)d_out;                 // [32,256,56,56]

    // workspace: rw (32x3 f32) at 0, mixed bf16 weights at +512 (37.75 MB)
    float* rw = (float*)d_ws;
    unsigned short* cwb = (unsigned short*)((char*)d_ws + 512);

    route_kernel<<<BATCH, 256, 0, stream>>>(x, rcw, rcb, fcw, fcb, rw);
    mix_kernel<<<dim3(COUT, NTAP, BATCH), 256, 0, stream>>>(we, rw, cwb);
    conv_kernel<<<dim3(49, COUT / MT, BATCH), CTHREADS, 0, stream>>>(x, cwb, out);
}
